// STFF_v2_39170101740110
// MI455X (gfx1250) — matrix operand order repaired and checked by reading, not compiled, not run on hardware
//
#include <hip/hip_runtime.h>
#include <hip/hip_bf16.h>

// ---------------------------------------------------------------------------
// CDNA5 / gfx1250 ConvLSTM implementation.
// Conv3x3 SAME -> implicit GEMM on v_wmma_f32_16x16x32_bf16 (wave32).
// NHWC bf16 hidden state, tap-major K layout -> contiguous 128-bit staging,
// double-buffered LDS pipeline (one barrier per K-chunk).
// ---------------------------------------------------------------------------

typedef __attribute__((ext_vector_type(16))) __bf16 v16bf;
typedef __attribute__((ext_vector_type(8)))  float  v8f;

#define HH 38
#define WW 30
#define PIX (HH*WW)          // 1140
#define BATCH 8
#define MROWS (BATCH*PIX)    // 9120
#define MPAD 9152            // MROWS rounded up to 64
#define STEPS 100
#define CINP 160             // padded concat channels (both layers)
#define KPAD (9*CINP)        // 1440, K = (kh*3+kw)*CINP + c
#define NCHUNK (KPAD/32)     // 45

// fp32 -> bf16 round-to-nearest-even without relying on native __bf16 convert
__device__ __forceinline__ __bf16 f2bf(float f) {
    unsigned u = __float_as_uint(f);
    u += 0x7fffu + ((u >> 16) & 1u);
    unsigned short s = (unsigned short)(u >> 16);
    __bf16 b;
    __builtin_memcpy(&b, &s, 2);
    return b;
}

__device__ __forceinline__ float sigf(float x) {
    return 1.0f / (1.0f + __expf(-x));
}

__device__ __forceinline__ float tanh_fast(float x) {
    x = fminf(fmaxf(x, -15.0f), 15.0f);
    float e = __expf(2.0f * x);
    return (e - 1.0f) / (e + 1.0f);
}

// ---------------------------------------------------------------------------
// Weight packing: bf16 B matrix [NTOT][KPAD], NTOT = 4*HID (gate order i,f,c,o)
// k = (kh*3+kw)*CINP + c ; c<CX from Wx (HID,CX,3,3), CX<=c<CX+HID from Wh.
// ---------------------------------------------------------------------------
template<int HID, int CX>
__global__ void stff_pack_kernel(const float* __restrict__ Wxi, const float* __restrict__ Wxf,
                                 const float* __restrict__ Wxc, const float* __restrict__ Wxo,
                                 const float* __restrict__ Whi, const float* __restrict__ Whf,
                                 const float* __restrict__ Whc, const float* __restrict__ Who,
                                 __bf16* __restrict__ Wp) {
    const int NTOT = 4 * HID;
    int idx = blockIdx.x * blockDim.x + threadIdx.x;
    if (idx >= NTOT * KPAD) return;
    int n = idx / KPAD;
    int k = idx - n * KPAD;
    int g = n / HID;
    int o = n - g * HID;
    int slab = k / CINP;          // kh*3+kw
    int c = k - slab * CINP;
    const float* Wx[4] = {Wxi, Wxf, Wxc, Wxo};
    const float* Wh[4] = {Whi, Whf, Whc, Who};
    float val = 0.0f;
    if (c < CX)                  val = Wx[g][(o * CX + c) * 9 + slab];
    else if (c < CX + HID)       val = Wh[g][(o * HID + (c - CX)) * 9 + slab];
    Wp[idx] = f2bf(val);
}

// ---------------------------------------------------------------------------
// Copy x_t into channel 0 of the NHWC bf16 concat buffer (8, 1140, 160)
// ---------------------------------------------------------------------------
__global__ void stff_copyx_kernel(const float* __restrict__ x, __bf16* __restrict__ cat1, int t) {
    int idx = blockIdx.x * blockDim.x + threadIdx.x;
    if (idx >= BATCH * PIX) return;
    int b = idx / PIX;
    int pix = idx - b * PIX;
    cat1[(size_t)idx * CINP] = f2bf(x[((size_t)b * STEPS + t) * PIX + pix]);
}

// ---------------------------------------------------------------------------
// Implicit-GEMM gate computation on WMMA bf16.
//   gates[MPAD][NTOT] = im2col(cat)[M][KPAD] * Wp^T   (Wp is [NTOT][KPAD])
// Block: 256 threads = 8 waves. Tile M64 x N128; wave owns 16x64 (4 n-subtiles).
// Flat K loop of 45 chunks, double-buffered LDS: stage chunk n+1 while
// computing chunk n; one barrier per chunk. A load is a clamped (always
// valid) 16B global load with data zero-select from a precomputed tap mask.
// ---------------------------------------------------------------------------
template<int NTOT>
__global__ __launch_bounds__(256)
void stff_gates_gemm_kernel(const __bf16* __restrict__ cat,   // (8, 1140, CINP) NHWC
                            const __bf16* __restrict__ Wp,    // (NTOT, KPAD)
                            float* __restrict__ gates) {      // (MPAD, NTOT)
    __shared__ __bf16 lA[2][64][32];    // A tiles (M x Kc)
    __shared__ __bf16 lB[2][128][32];   // B tiles (N x Kc)

    const int tid  = threadIdx.x;
    const int wave = tid >> 5;
    const int lane = tid & 31;
    const int mBase = blockIdx.y * 64;
    const int nBase = blockIdx.x * 128;

    const int mSub  = wave & 3;        // rows mSub*16
    const int nHalf = wave >> 2;       // cols nHalf*64

    v8f acc[4] = {};

    // A-staging coords for this thread (8 bf16 of one row per chunk)
    const int fillIdx = tid * 8;            // 0..2047
    const int fRow = fillIdx >> 5;          // 0..63
    const int fKb  = fillIdx & 31;          // 0,8,16,24
    const int fM   = mBase + fRow;
    const int fB   = fM / PIX;
    const int fPix = fM - fB * PIX;
    const int fOh  = fPix / WW;
    const int fOw  = fPix - fOh * WW;

    // per-tap in-bounds mask (9 bits)
    unsigned inbMask = 0;
    if (fM < MROWS) {
        #pragma unroll
        for (int s = 0; s < 9; ++s) {
            int kh = s / 3, kw = s % 3;               // compile-time
            int ih = fOh + kh - 1, iw = fOw + kw - 1;
            if ((unsigned)ih < (unsigned)HH && (unsigned)iw < (unsigned)WW)
                inbMask |= 1u << s;
        }
    }

    // B-staging coords: one n row, 16 bf16 (two float4)
    const int bN   = tid >> 1;              // 0..127
    const int bKo  = (tid & 1) * 16;        // 0 or 16
    const __bf16* wRow = &Wp[(size_t)(nBase + bN) * KPAD];

    // fragment-load constants
    const int hsel = lane >> 4;
    const int aRow = (mSub << 4) + (lane & 15);

    const float4 zero4 = {0.0f, 0.0f, 0.0f, 0.0f};

    // ---- staging helpers ----
    auto stageA = [&](int chunk, int buf) {
        int slab = chunk / 5;                    // 0..8
        int cc   = (chunk - slab * 5) * 32;      // 0..128
        // dpix = (kh-1)*WW + (kw-1) = slab + 27*(slab/3) - 31
        int kh   = slab / 3;
        int dpix = slab + 27 * kh - 31;
        int pc   = fM + dpix;                    // flat pixel index in cat
        pc = pc < 0 ? 0 : (pc >= MROWS ? MROWS - 1 : pc);   // clamp: always valid
        float4 av = *reinterpret_cast<const float4*>(
            cat + (size_t)pc * CINP + cc + fKb);
        if (!((inbMask >> slab) & 1u)) av = zero4;          // data select, not addr
        *reinterpret_cast<float4*>(&lA[buf][fRow][fKb]) = av;
    };
    auto stageB = [&](int chunk, int buf) {
        const float4* src = reinterpret_cast<const float4*>(wRow + chunk * 32 + bKo);
        float4* dst = reinterpret_cast<float4*>(&lB[buf][bN][bKo]);
        dst[0] = src[0];
        dst[1] = src[1];
    };

    stageA(0, 0);
    stageB(0, 0);

    for (int chunk = 0; chunk < NCHUNK; ++chunk) {
        __syncthreads();                 // staged data for `chunk` visible
        const int cb = chunk & 1;
        if (chunk + 1 < NCHUNK) {
            stageA(chunk + 1, cb ^ 1);
            stageB(chunk + 1, cb ^ 1);
            if (chunk + 2 < NCHUNK)
                __builtin_prefetch(wRow + (chunk + 2) * 32, 0, 3);  // global_prefetch_b8
        }

        // ---- A fragment: lane<16 -> K {0..7,16..23}; lane>=16 -> {8..15,24..31}
        union { v16bf v; float4 q[2]; } af;
        const float4* ar = reinterpret_cast<const float4*>(&lA[cb][aRow][0]);
        af.q[0] = ar[hsel];
        af.q[1] = ar[hsel + 2];

        // ---- 4 WMMA per chunk (one per n-subtile) ----
        #pragma unroll
        for (int j = 0; j < 4; ++j) {
            const int col = nHalf * 64 + j * 16 + (lane & 15);
            union { v16bf v; float4 q[2]; } bfm;
            const float4* br = reinterpret_cast<const float4*>(&lB[cb][col][0]);
            bfm.q[0] = br[hsel];
            bfm.q[1] = br[hsel + 2];
            acc[j] = __builtin_amdgcn_wmma_f32_16x16x32_bf16(
                false, af.v, false, bfm.v, (short)0, acc[j], false, false);
        }
    }

    // ---- store (unconditional: gates padded to MPAD rows)
    const int rowBase = mBase + (mSub << 4) + (hsel << 3);
    #pragma unroll
    for (int j = 0; j < 4; ++j) {
        const int col = nBase + nHalf * 64 + j * 16 + (lane & 15);
        #pragma unroll
        for (int v = 0; v < 8; ++v) {
            gates[(size_t)(rowBase + v) * NTOT + col] = acc[j][v];
        }
    }
}

// ---------------------------------------------------------------------------
// Layer-1 LSTM elementwise update (hid=128). Writes c1 (f32) and h1 as bf16
// into cat1 (channel 1+o) and cat2 (channel o), NHWC.
// ---------------------------------------------------------------------------
__global__ void stff_cell1_kernel(const float* __restrict__ gates,   // (MPAD, 512)
                                  float* __restrict__ c1,            // (8,128,1140)
                                  __bf16* __restrict__ cat1,         // (8,1140,160)
                                  __bf16* __restrict__ cat2,         // (8,1140,160)
                                  const float* __restrict__ bxi, const float* __restrict__ bxf,
                                  const float* __restrict__ bxc, const float* __restrict__ bxo,
                                  const float* __restrict__ Wci, const float* __restrict__ Wcf,
                                  const float* __restrict__ Wco) {
    int idx = blockIdx.x * blockDim.x + threadIdx.x;
    if (idx >= BATCH * 128 * PIX) return;
    int b = idx / (128 * PIX);
    int rem = idx - b * 128 * PIX;
    int o = rem / PIX;
    int pix = rem - o * PIX;
    size_t m = (size_t)b * PIX + pix;
    const float* G = gates + m * 512;
    float c = c1[idx];
    float ig = sigf(G[o]             + bxi[o] + c * Wci[o * PIX + pix]);
    float fg = sigf(G[128 + o]       + bxf[o] + c * Wcf[o * PIX + pix]);
    float gg = tanh_fast(G[256 + o]  + bxc[o]);
    float cc = fg * c + ig * gg;
    float og = sigf(G[384 + o]       + bxo[o] + cc * Wco[o * PIX + pix]);
    float h  = og * tanh_fast(cc);
    c1[idx] = cc;
    __bf16 hb = f2bf(h);
    cat1[m * CINP + 1 + o] = hb;
    cat2[m * CINP + o]     = hb;
}

// ---------------------------------------------------------------------------
// Layer-2 LSTM elementwise update (hid=32). Writes c2, h2 (f32 for decoder)
// and h2 bf16 into cat2 (channel 128+o), NHWC.
// ---------------------------------------------------------------------------
__global__ void stff_cell2_kernel(const float* __restrict__ gates,   // (MPAD, 128)
                                  float* __restrict__ c2,            // (8,32,1140)
                                  float* __restrict__ h2f,           // (8,32,1140)
                                  __bf16* __restrict__ cat2,         // (8,1140,160)
                                  const float* __restrict__ bxi, const float* __restrict__ bxf,
                                  const float* __restrict__ bxc, const float* __restrict__ bxo,
                                  const float* __restrict__ Wci, const float* __restrict__ Wcf,
                                  const float* __restrict__ Wco) {
    int idx = blockIdx.x * blockDim.x + threadIdx.x;
    if (idx >= BATCH * 32 * PIX) return;
    int b = idx / (32 * PIX);
    int rem = idx - b * 32 * PIX;
    int o = rem / PIX;
    int pix = rem - o * PIX;
    size_t m = (size_t)b * PIX + pix;
    const float* G = gates + m * 128;
    float c = c2[idx];
    float ig = sigf(G[o]            + bxi[o] + c * Wci[o * PIX + pix]);
    float fg = sigf(G[32 + o]       + bxf[o] + c * Wcf[o * PIX + pix]);
    float gg = tanh_fast(G[64 + o]  + bxc[o]);
    float cc = fg * c + ig * gg;
    float og = sigf(G[96 + o]       + bxo[o] + cc * Wco[o * PIX + pix]);
    float h  = og * tanh_fast(cc);
    c2[idx] = cc;
    h2f[idx] = h;
    cat2[m * CINP + 128 + o] = f2bf(h);
}

// ---------------------------------------------------------------------------
// Decoder: conv_transpose 32->16, 3x3, stride 2, VALID (out 77x61) + ReLU
// ---------------------------------------------------------------------------
__global__ void stff_dec1_kernel(const float* __restrict__ h2f,  // (8,32,38,30)
                                 const float* __restrict__ Wt1,  // (32,16,3,3)
                                 const float* __restrict__ bt1,  // (16)
                                 float* __restrict__ d1) {       // (8,16,77,61)
    int idx = blockIdx.x * blockDim.x + threadIdx.x;
    const int OH = 77, OW = 61;
    if (idx >= BATCH * 16 * OH * OW) return;
    int b = idx / (16 * OH * OW);
    int rem = idx - b * 16 * OH * OW;
    int oc = rem / (OH * OW);
    rem -= oc * OH * OW;
    int y = rem / OW;
    int x = rem - y * OW;
    float acc = bt1[oc];
    for (int kh = 0; kh < 3; ++kh) {
        int iy = y - kh;
        if (iy < 0 || (iy & 1)) continue;
        iy >>= 1;
        if (iy >= HH) continue;
        for (int kw = 0; kw < 3; ++kw) {
            int ix = x - kw;
            if (ix < 0 || (ix & 1)) continue;
            ix >>= 1;
            if (ix >= WW) continue;
            for (int ic = 0; ic < 32; ++ic) {
                acc += h2f[((size_t)(b * 32 + ic) * HH + iy) * WW + ix] *
                       Wt1[((ic * 16) + oc) * 9 + kh * 3 + kw];
            }
        }
    }
    d1[idx] = fmaxf(acc, 0.0f);
}

// ---------------------------------------------------------------------------
// Decoder: conv_transpose 16->1, 5x5, stride 2, VALID (out 157x125)
// ---------------------------------------------------------------------------
__global__ void stff_dec2_kernel(const float* __restrict__ d1,   // (8,16,77,61)
                                 const float* __restrict__ Wt2,  // (16,1,5,5)
                                 const float* __restrict__ bt2,  // (1)
                                 float* __restrict__ dd) {       // (8,157,125)
    int idx = blockIdx.x * blockDim.x + threadIdx.x;
    const int IH = 77, IW = 61, OH = 157, OW = 125;
    if (idx >= BATCH * OH * OW) return;
    int b = idx / (OH * OW);
    int rem = idx - b * OH * OW;
    int y = rem / OW;
    int x = rem - y * OW;
    float acc = bt2[0];
    for (int kh = 0; kh < 5; ++kh) {
        int iy = y - kh;
        if (iy < 0 || (iy & 1)) continue;
        iy >>= 1;
        if (iy >= IH) continue;
        for (int kw = 0; kw < 5; ++kw) {
            int ix = x - kw;
            if (ix < 0 || (ix & 1)) continue;
            ix >>= 1;
            if (ix >= IW) continue;
            for (int ic = 0; ic < 16; ++ic) {
                acc += d1[((size_t)(b * 16 + ic) * IH + iy) * IW + ix] *
                       Wt2[ic * 25 + kh * 5 + kw];
            }
        }
    }
    dd[idx] = acc;
}

// ---------------------------------------------------------------------------
// Interval-average pooling 157->135 (rows), 125->103 (cols) == MH @ out @ MW^T
// ---------------------------------------------------------------------------
__global__ void stff_pool_kernel(const float* __restrict__ dd,  // (8,157,125)
                                 float* __restrict__ out) {     // (8,135,103)
    int idx = blockIdx.x * blockDim.x + threadIdx.x;
    const int IH = 157, IW = 125, OH = 135, OW = 103;
    if (idx >= BATCH * OH * OW) return;
    int b = idx / (OH * OW);
    int rem = idx - b * OH * OW;
    int o = rem / OW;
    int p = rem - o * OW;
    int sh = (o * IH) / OH;
    int eh = ((o + 1) * IH + OH - 1) / OH;
    int sw = (p * IW) / OW;
    int ew = ((p + 1) * IW + OW - 1) / OW;
    float sum = 0.0f;
    for (int h = sh; h < eh; ++h)
        for (int w = sw; w < ew; ++w)
            sum += dd[((size_t)b * IH + h) * IW + w];
    out[idx] = sum / ((float)(eh - sh) * (float)(ew - sw));
}

// ---------------------------------------------------------------------------
// Host launch
// ---------------------------------------------------------------------------
extern "C" void kernel_launch(void* const* d_in, const int* in_sizes, int n_in,
                              void* d_out, int out_size, void* d_ws, size_t ws_size,
                              hipStream_t stream) {
    (void)in_sizes; (void)n_in; (void)out_size; (void)ws_size;

    // Input order: x, then params1/params2 in insertion order
    // (Wxi,Wxf,Wxc,Wxo,Whi,Whf,Whc,Who,Wci,Wcf,Wco,bxi,bxf,bxc,bxo), then dec.
    const float* x = (const float*)d_in[0];
    const float* p1[15]; for (int i = 0; i < 15; ++i) p1[i] = (const float*)d_in[1 + i];
    const float* p2[15]; for (int i = 0; i < 15; ++i) p2[i] = (const float*)d_in[16 + i];
    const float* Wt1 = (const float*)d_in[31];
    const float* bt1 = (const float*)d_in[32];
    const float* Wt2 = (const float*)d_in[33];
    const float* bt2 = (const float*)d_in[34];
    float* out = (float*)d_out;

    // Workspace carve-up (256B aligned)
    char* ws = (char*)d_ws;
    size_t off = 0;
    auto carve = [&](size_t bytes) -> char* {
        char* p = ws + off;
        off = (off + bytes + 255) & ~(size_t)255;
        return p;
    };
    // --- state (zero-initialized) ---
    float*  c1   = (float*) carve((size_t)BATCH * 128 * PIX * 4);
    float*  c2   = (float*) carve((size_t)BATCH * 32  * PIX * 4);
    __bf16* cat1 = (__bf16*)carve((size_t)BATCH * PIX * CINP * 2);  // NHWC
    __bf16* cat2 = (__bf16*)carve((size_t)BATCH * PIX * CINP * 2);  // NHWC
    float*  h2f  = (float*) carve((size_t)BATCH * 32  * PIX * 4);
    size_t stateBytes = off;
    // --- scratch ---
    __bf16* W1p = (__bf16*)carve((size_t)512 * KPAD * 2);
    __bf16* W2p = (__bf16*)carve((size_t)128 * KPAD * 2);
    float*  g1  = (float*) carve((size_t)MPAD * 512 * 4);
    float*  g2  = (float*) carve((size_t)MPAD * 128 * 4);
    float*  d1b = (float*) carve((size_t)BATCH * 16 * 77 * 61 * 4);
    float*  ddb = (float*) carve((size_t)BATCH * 157 * 125 * 4);

    hipMemsetAsync(d_ws, 0, stateBytes, stream);

    // Pack weights -> bf16 [N][KPAD]
    {
        int n1 = 512 * KPAD;
        stff_pack_kernel<128, 1><<<(n1 + 255) / 256, 256, 0, stream>>>(
            p1[0], p1[1], p1[2], p1[3], p1[4], p1[5], p1[6], p1[7], W1p);
        int n2 = 128 * KPAD;
        stff_pack_kernel<32, 128><<<(n2 + 255) / 256, 256, 0, stream>>>(
            p2[0], p2[1], p2[2], p2[3], p2[4], p2[5], p2[6], p2[7], W2p);
    }

    dim3 blk(256);
    dim3 grid1(4, MPAD / 64);   // 512/128 N-tiles x 143 M-tiles
    dim3 grid2(1, MPAD / 64);

    for (int t = 0; t < STEPS; ++t) {
        stff_copyx_kernel<<<(BATCH * PIX + 255) / 256, blk, 0, stream>>>(x, cat1, t);

        stff_gates_gemm_kernel<512><<<grid1, blk, 0, stream>>>(cat1, W1p, g1);
        stff_cell1_kernel<<<(BATCH * 128 * PIX + 255) / 256, blk, 0, stream>>>(
            g1, c1, cat1, cat2,
            p1[11], p1[12], p1[13], p1[14],   // bxi,bxf,bxc,bxo
            p1[8], p1[9], p1[10]);            // Wci,Wcf,Wco

        stff_gates_gemm_kernel<128><<<grid2, blk, 0, stream>>>(cat2, W2p, g2);
        stff_cell2_kernel<<<(BATCH * 32 * PIX + 255) / 256, blk, 0, stream>>>(
            g2, c2, h2f, cat2,
            p2[11], p2[12], p2[13], p2[14],
            p2[8], p2[9], p2[10]);
    }

    stff_dec1_kernel<<<(BATCH * 16 * 77 * 61 + 255) / 256, blk, 0, stream>>>(h2f, Wt1, bt1, d1b);
    stff_dec2_kernel<<<(BATCH * 157 * 125 + 255) / 256, blk, 0, stream>>>(d1b, Wt2, bt2, ddb);
    stff_pool_kernel<<<(BATCH * 135 * 103 + 255) / 256, blk, 0, stream>>>(ddb, out);
}
